// SpikingReservoirLoaded_6459630813514
// MI455X (gfx1250) — matrix operand order repaired and checked by reading, not compiled, not run on hardware
//
#include <hip/hip_runtime.h>

// ---------------- problem constants ----------------
#define RSZ      2048          // reservoir size R
#define BSZ      64            // batch
#define TSTEPS   500
#define BETA     0.9f
#define THRESH   1.0f

// ---------------- launch config --------------------
#define NBLOCKS  128           // one N-tile per block (grid-sync hardcodes this)
#define NTHREADS 512           // 16 waves: 4 mtiles x 4 ksplits

typedef __bf16 bf16;
typedef __attribute__((ext_vector_type(16))) __bf16 v16bf;
typedef __attribute__((ext_vector_type(8)))  float  v8f;

union Frag { uint4 u[2]; v16bf v; };   // 32 bytes: one WMMA A/B fragment per lane

// ------------- device-wide generation barrier -------------
__device__ __forceinline__ void grid_sync(unsigned* counter, unsigned* gen) {
  __syncthreads();
  if (threadIdx.x == 0) {
    __threadfence();  // make this block's S stores visible at device scope
    unsigned g = __hip_atomic_load(gen, __ATOMIC_RELAXED, __HIP_MEMORY_SCOPE_AGENT);
    unsigned a = __hip_atomic_fetch_add(counter, 1u, __ATOMIC_ACQ_REL, __HIP_MEMORY_SCOPE_AGENT);
    if (a == NBLOCKS - 1u) {
      __hip_atomic_store(counter, 0u, __ATOMIC_RELAXED, __HIP_MEMORY_SCOPE_AGENT);
      __hip_atomic_store(gen, g + 1u, __ATOMIC_RELEASE, __HIP_MEMORY_SCOPE_AGENT);
    } else {
      while (__hip_atomic_load(gen, __ATOMIC_ACQUIRE, __HIP_MEMORY_SCOPE_AGENT) == g) {
        __builtin_amdgcn_s_sleep(2);
      }
    }
  }
  __syncthreads();
}

// ------------- init: barrier, S0 = 0, out[0] = 0, transpose x -> xT (T,B) ---
__global__ void sr_init_kernel(const float* __restrict__ x, float* out,
                               unsigned* bar, bf16* sbuf, float* xT) {
  unsigned tid = blockIdx.x * blockDim.x + threadIdx.x;
  if (tid < (unsigned)(BSZ * RSZ)) sbuf[tid] = (bf16)0.0f;  // ping buffer 0
  if (tid < (unsigned)(BSZ * TSTEPS)) {                     // xT[t*B+b]=x[b*T+t]
    unsigned b = tid & 63u, t = tid >> 6;
    xT[tid] = x[(size_t)b * TSTEPS + t];
  }
  if (tid == 0) { bar[0] = 0u; bar[1] = 0u; out[0] = 0.0f; }
}

// ------------- pack W (f32, row-major R x R) into bf16 hi/lo B-fragments -----
// element e (0..15) of lane L holds W[k, n] with
//   k = kk64*32 + (L<16 ? 0 : 16) + e,   n = ntile*16 + (L&15)
// flat index = ((kk64*128 + ntile)*32 + L)*16 + e
__global__ void sr_pack_kernel(const float* __restrict__ W,
                               bf16* __restrict__ Wh, bf16* __restrict__ Wl) {
  unsigned tid   = blockIdx.x * 256u + threadIdx.x;   // 0 .. R*R-1 (4M)
  unsigned e     = tid & 15u;
  unsigned lane  = (tid >> 4) & 31u;
  unsigned ntile = (tid >> 9) & 127u;
  unsigned kk64  = tid >> 16;
  unsigned k = kk64 * 32u + ((lane < 16u) ? 0u : 16u) + e;
  unsigned n = ntile * 16u + (lane & 15u);
  float w  = W[(size_t)k * RSZ + n];
  bf16 h   = (bf16)w;
  float lo = w - (float)h;
  Wh[tid] = h;
  Wl[tid] = (bf16)lo;
}

// ------------- persistent reservoir kernel -------------
// Block = one N-tile (16 cols). 16 waves = 4 mtiles x 4 ksplits.
// Full hi/lo B panel (128 KB) staged in LDS once (W constant over time).
// LIF update + record stores are spread across ALL 16 waves (2 elems/lane).
__global__ __launch_bounds__(NTHREADS)
void sr_reservoir_kernel(const float* __restrict__ xT,     // (T, B) transposed
                         const float* __restrict__ w_in,   // (R, 1)
                         const bf16*  __restrict__ Wh,
                         const bf16*  __restrict__ Wl,
                         bf16* __restrict__ sbuf,          // 2 * B * R (ping-pong)
                         unsigned* bar,
                         float* __restrict__ out)          // [avg][spk TBR][mem TBR]
{
  __shared__ float xs[BSZ];
  __shared__ float red[16][8][32];                         // [wave][r][lane], 16 KB
  __shared__ __align__(16) unsigned char bsh[64 * 2048];   // 128 KB B panel (hi+lo)

  const unsigned tid     = threadIdx.x;
  const unsigned lane    = tid & 31u;
  const unsigned wave    = tid >> 5;        // 0..15
  const unsigned ksplit  = wave & 3u;       // 0..3
  const unsigned ntile   = blockIdx.x;      // 0..127
  const bool     hiHalf  = lane >= 16u;
  const unsigned arow    = (wave >> 2) * 16u + (lane & 15u);  // A row (mtile based)
  const unsigned kk64base = ksplit * 16u;   // 16 chunks of K=32 per wave

  // ---- one-time stage of B panel (hi+lo, all K) into LDS ----
  // LDS layout: byte offset = kk64*2048 + mat*1024 + half*512 + lane*16
  for (unsigned idx = wave; idx < 128u; idx += 16u) {
    unsigned kk64 = idx >> 1, mat = idx & 1u;
    const bf16* src = (mat ? Wl : Wh) +
                      ((size_t)(kk64 * 128u + ntile) * 32u + lane) * 16u;
    uint4 h0 = *(const uint4*)src;
    uint4 h1 = *(const uint4*)(src + 8);
    unsigned off = kk64 * 2048u + mat * 1024u + lane * 16u;
    *(uint4*)(bsh + off)        = h0;
    *(uint4*)(bsh + off + 512u) = h1;
  }

  // ---- per-lane LIF-element assignment: eid = wave*64 + j*32 + lane ----
  unsigned bj[2], rj[2], lredj[2], w0j[2];
  float winj[2];
  size_t offj[2];
  #pragma unroll
  for (int j = 0; j < 2; ++j) {
    unsigned eid  = wave * 64u + (unsigned)j * 32u + lane;  // 0..1023
    unsigned b    = eid >> 4;            // batch row 0..63
    unsigned ncol = eid & 15u;           // column in N-tile
    unsigned rit  = b & 15u;             // row within 16x16 tile
    bj[j]    = b;
    rj[j]    = rit & 7u;                 // acc VGPR index
    lredj[j] = ((rit & 8u) ? 16u : 0u) + ncol;   // lane slot in red
    w0j[j]   = (b >> 4) * 4u;            // first wave of this mtile group
    winj[j]  = w_in[ntile * 16u + ncol];
    offj[j]  = (size_t)b * RSZ + ntile * 16u + ncol;
  }

  float* spike_rec = out + 1;
  float* mem_rec   = out + 1 + (size_t)TSTEPS * BSZ * RSZ;

  float V[2] = {0.0f, 0.0f};
  float spikes = 0.0f;

  __syncthreads();   // B panel ready

  for (int t = 0; t < TSTEPS; ++t) {
    if (tid < BSZ) xs[tid] = xT[(size_t)t * BSZ + tid];   // coalesced
    __syncthreads();

    const bf16* Sprev = sbuf + (size_t)(t & 1) * (BSZ * RSZ);
    bf16*       Snext = sbuf + (size_t)((t + 1) & 1) * (BSZ * RSZ);

    // two independent accumulator chains (hi / lo) for WMMA ILP
    v8f acch = {0.f, 0.f, 0.f, 0.f, 0.f, 0.f, 0.f, 0.f};
    v8f accl = {0.f, 0.f, 0.f, 0.f, 0.f, 0.f, 0.f, 0.f};
    const bf16* arowp = Sprev + (size_t)arow * RSZ;

    #pragma unroll 2
    for (unsigned i = 0; i < 16u; ++i) {
      const unsigned kk64 = kk64base + i;
      const unsigned kk   = kk64 * 32u;
      Frag A, Bh, Bl;
      // A: contiguous K order (not ISA table order) — low lanes K {0..15}, high lanes {16..31} (matches B pack)
      A.u[0] = *(const uint4*)(arowp + kk + (hiHalf ? 16u : 0u));
      A.u[1] = *(const uint4*)(arowp + kk + (hiHalf ? 24u : 8u));
      const unsigned boff = kk64 * 2048u + lane * 16u;
      Bh.u[0] = *(const uint4*)(bsh + boff);
      Bh.u[1] = *(const uint4*)(bsh + boff + 512u);
      Bl.u[0] = *(const uint4*)(bsh + boff + 1024u);
      Bl.u[1] = *(const uint4*)(bsh + boff + 1536u);
      acch = __builtin_amdgcn_wmma_f32_16x16x32_bf16(false, A.v, false, Bh.v,
                                                     (short)0, acch, false, false);
      accl = __builtin_amdgcn_wmma_f32_16x16x32_bf16(false, A.v, false, Bl.v,
                                                     (short)0, accl, false, false);
    }

    // cross-wave K reduction through LDS (conflict-free: lanes contiguous per r)
    #pragma unroll
    for (int r = 0; r < 8; ++r) red[wave][r][lane] = acch[r] + accl[r];
    __syncthreads();

    // LIF update spread over all 16 waves: 2 elements per lane
    #pragma unroll
    for (int j = 0; j < 2; ++j) {
      float a = red[w0j[j] + 0][rj[j]][lredj[j]] + red[w0j[j] + 1][rj[j]][lredj[j]] +
                red[w0j[j] + 2][rj[j]][lredj[j]] + red[w0j[j] + 3][rj[j]][lredj[j]];
      float Vn = BETA * V[j] + xs[bj[j]] * winj[j] + a;
      float S  = (Vn >= THRESH) ? 1.0f : 0.0f;
      spikes  += S;
      float Vr = Vn * (1.0f - S);            // zero reset
      V[j] = Vr;
      size_t off = (size_t)t * (BSZ * RSZ) + offj[j];
      __builtin_nontemporal_store(S,  spike_rec + off);   // streaming stores
      __builtin_nontemporal_store(Vr, mem_rec   + off);
      Snext[offj[j]] = (bf16)S;                           // re-read next step
    }
    grid_sync(bar, bar + 1);
  }

  const float scale = 1.0f / ((float)TSTEPS * (float)BSZ * (float)RSZ);
  atomicAdd(out, spikes * scale);
}

// ------------- launch -------------
extern "C" void kernel_launch(void* const* d_in, const int* in_sizes, int n_in,
                              void* d_out, int out_size, void* d_ws, size_t ws_size,
                              hipStream_t stream) {
  (void)in_sizes; (void)n_in; (void)out_size; (void)ws_size;
  const float* x    = (const float*)d_in[0];   // (64, 500, 1)
  const float* w_in = (const float*)d_in[1];   // (2048, 1)
  const float* W    = (const float*)d_in[2];   // (2048, 2048)
  float* out = (float*)d_out;

  // workspace: barrier | S ping-pong (512 KB) | Wh (8 MB) | Wl (8 MB) | xT (125 KB)
  char* ws = (char*)d_ws;
  unsigned* bar = (unsigned*)ws;
  bf16*  sbuf = (bf16*)(ws + 256);
  bf16*  Wh   = (bf16*)(ws + 256 + (size_t)2 * BSZ * RSZ * 2);
  bf16*  Wl   = Wh + (size_t)RSZ * RSZ;
  float* xT   = (float*)(ws + 256 + (size_t)2 * BSZ * RSZ * 2 +
                         (size_t)2 * RSZ * RSZ * 2);

  sr_init_kernel<<<(BSZ * RSZ) / 256, 256, 0, stream>>>(x, out, bar, sbuf, xT);
  sr_pack_kernel<<<(RSZ * RSZ) / 256, 256, 0, stream>>>(W, Wh, Wl);
  sr_reservoir_kernel<<<NBLOCKS, NTHREADS, 0, stream>>>(xT, w_in, Wh, Wl, sbuf, bar, out);
}
